// GraphRestrictedBoltzmannMachine_64682207478108
// MI455X (gfx1250) — matrix operand order repaired and checked by reading, not compiled, not run on hardware
//
#include <hip/hip_runtime.h>
#include <hip/hip_bf16.h>
#include <stdint.h>

// ---------------------------------------------------------------------------
// Graph-restricted Boltzmann machine hidden-field kernel for gfx1250.
//
// field[k,b] = sum_{e: seg_ids[e]==k} quad[jidx[e]] * s_pm[b, adj[e]]
// out[b,k]   = tanh(-(field[k,b] + linear[hidden_idx[k]]))
//
// s_pm in {-1,+1} is bit-packed to 128 bits/node (880 KB, L2 resident).
// CSR build: per-block LDS histograms (low global-atomic pressure), then a
// single-wave WMMA prefix scan (inclusive scan == triangular-matrix product,
// exact in f16->f32 for Poisson(200) bin counts), then LDS-reserved scatter.
// Main kernel: 1 block (4 wave32s) per hidden unit; edge (v,w) pairs are
// double-buffered into LDS with GLOBAL_LOAD_ASYNC_TO_LDS_B64 (ASYNCcnt),
// then each (edge,batch) pair is a sign-xor accumulate.
// ---------------------------------------------------------------------------

#define CSR_BLOCKS 80
#define EDGE_TILE  128

typedef __attribute__((ext_vector_type(16))) _Float16 v16h;
typedef __attribute__((ext_vector_type(8)))  float    v8f;

// ----- gfx1250 async global->LDS helpers (ASYNCcnt-tracked DMA) ------------
__device__ __forceinline__ void async_load_edge(uint32_t lds_byte_addr,
                                                uint32_t byte_off,
                                                const uint2* base) {
    // GVS mode: mem = SGPR64(base) + VGPR32(byte_off); LDS dest from VGPR.
    asm volatile("global_load_async_to_lds_b64 %0, %1, %2"
                 :
                 : "v"(lds_byte_addr), "v"(byte_off), "s"(base)
                 : "memory");
}
__device__ __forceinline__ void async_wait_0() {
    asm volatile("s_wait_asynccnt 0x0" ::: "memory");
}
__device__ __forceinline__ void async_wait_le1() {
    asm volatile("s_wait_asynccnt 0x1" ::: "memory");
}

// ---------------------------------------------------------------------------
__global__ void zero_u32_kernel(uint32_t* __restrict__ p, int n) {
    int i = blockIdx.x * blockDim.x + threadIdx.x;
    if (i < n) p[i] = 0u;
}

// Pack s (B x N, {0,1} int32) into per-node bitmasks: s_bits[v*NW + w] holds
// batch bits b = 32w .. 32w+31. Loads coalesce across consecutive v lanes.
__global__ void pack_bits_kernel(const int* __restrict__ s,
                                 uint32_t* __restrict__ s_bits,
                                 int N, int NW) {
    int v = blockIdx.x * blockDim.x + threadIdx.x;
    if (v >= N) return;
    for (int w = 0; w < NW; ++w) {
        uint32_t word = 0u;
        #pragma unroll 8
        for (int i = 0; i < 32; ++i) {
            word |= (uint32_t)(s[(size_t)(w * 32 + i) * N + v] & 1) << i;
        }
        s_bits[(size_t)v * NW + w] = word;
    }
}

// Per-block LDS histogram of segment ids, merged with one global atomic per
// (block, nonzero bin): ~CSR_BLOCKS*K uncontended atomics instead of E.
__global__ void count_lds_kernel(const int* __restrict__ seg_ids,
                                 uint32_t* __restrict__ counts, int E, int K) {
    extern __shared__ uint32_t h_cnt[];
    const int t = threadIdx.x;
    for (int i = t; i < K; i += blockDim.x) h_cnt[i] = 0u;
    __syncthreads();
    for (int e = blockIdx.x * blockDim.x + t; e < E; e += gridDim.x * blockDim.x)
        atomicAdd(&h_cnt[seg_ids[e]], 1u);
    __syncthreads();
    for (int i = t; i < K; i += blockDim.x) {
        uint32_t c = h_cnt[i];
        if (c) atomicAdd(&counts[i], c);
    }
}

// ---------------------------------------------------------------------------
// Single-wave WMMA prefix scan. Inclusive scan over 32-deep chunks is a
// matrix product with a constant lower-triangular ones matrix:
//   D(m,n) = sum_j A(m,j) * B(j,n),  A(m,j) = (j <= m+off),  B(j,n) = x[n*32+j]
// Two WMMAs (off=0 / off=16) produce all 32 scan positions for 16 chunks at
// once; 4 WMMAs cover 1024 bins. Exact: counts <= 2048 are exact f16, A is
// 0/1, accumulation in f32 (integers < 2^24 exact). Chunk carries are folded
// sequentially; a running carry chains the 1024-bin passes.
//
// VGPR layouts (ISA 7.12.2, wave32):
//   A 16x32 f16 (k = 16h+s here, not ISA table order): lane L<16 -> row L, halves = K 0..15;
//                lane L>=16 -> row L-16, halves = K 16..31
//   B 32x16 f16 (same k as A): lane L<16 -> col L, halves = K 0..15; L>=16 -> col L-16, K 16..31
//   D 16x16 f32: reg r, lane L<16 -> (M=r, N=L); lane L>=16 -> (M=r+8, N=L-16)
// ---------------------------------------------------------------------------
__global__ __launch_bounds__(32) void scan_counts_wmma_kernel(
    const uint32_t* __restrict__ counts, uint32_t* __restrict__ row_ptr, int K) {
    __shared__ __align__(32) _Float16 x16[1024];
    __shared__ float y[1024];
    __shared__ float carry_s[32];
    __shared__ float gcarry_s;
    const int lane = threadIdx.x;
    if (lane == 0) { gcarry_s = 0.0f; row_ptr[0] = 0u; }

    // Constant triangular A operands (built once per lane).
    v16h a1, a2;
    const int m     = lane & 15;
    const int kbase = (lane < 16) ? 0 : 16;
    #pragma unroll
    for (int i = 0; i < 16; ++i) {
        const int kk = kbase + i;
        a1[i] = (kk <= m)      ? (_Float16)1.0f : (_Float16)0.0f;
        a2[i] = (kk <= m + 16) ? (_Float16)1.0f : (_Float16)0.0f;
    }

    for (int base = 0; base < K; base += 1024) {
        // Stage 1024 bin counts as f16 (zero-padded past K).
        for (int i = lane; i < 1024; i += 32) {
            const int g = base + i;
            const uint32_t c = (g < K) ? counts[g] : 0u;
            x16[i] = (_Float16)(float)c;
        }
        __syncthreads();

        // Two half-passes: chunks 0..15 then 16..31.
        #pragma unroll
        for (int hp = 0; hp < 2; ++hp) {
            const int chunk = hp * 16 + (lane & 15);
            const _Float16* bp = &x16[chunk * 32 + ((lane < 16) ? 0 : 16)];
            const v16h b = *(const v16h*)bp;
            v8f c0 = {};
            v8f d1 = __builtin_amdgcn_wmma_f32_16x16x32_f16(
                false, a1, false, b, (short)0, c0, false, false);
            v8f d2 = __builtin_amdgcn_wmma_f32_16x16x32_f16(
                false, a2, false, b, (short)0, c0, false, false);
            float* yp = &y[chunk * 32 + ((lane < 16) ? 0 : 8)];
            #pragma unroll
            for (int r = 0; r < 8; ++r) yp[r] = d1[r];        // positions off+0..7
            #pragma unroll
            for (int r = 0; r < 8; ++r) yp[16 + r] = d2[r];   // positions off+16..23
        }
        __syncthreads();

        // Cross-chunk carries (32 f32 totals, sequential = exact order).
        if (lane == 0) {
            float run = gcarry_s;
            for (int c = 0; c < 32; ++c) { carry_s[c] = run; run += y[c * 32 + 31]; }
            gcarry_s = run;
        }
        __syncthreads();

        for (int i = lane; i < 1024; i += 32) {
            const int g = base + i;
            if (g < K) row_ptr[g + 1] = (uint32_t)(y[i] + carry_s[i >> 5]);
        }
        __syncthreads();
    }
}

__global__ void init_cursor_kernel(const uint32_t* __restrict__ row_ptr,
                                   uint32_t* __restrict__ cursor, int K) {
    int i = blockIdx.x * blockDim.x + threadIdx.x;
    if (i < K) cursor[i] = row_ptr[i];
}

// Scatter edges into CSR order. Each block rebuilds its LDS histogram,
// reserves a contiguous range per bin with ONE global atomic, then hands out
// positions within the block via LDS atomics. Edge (v, w-bits) interleaved.
__global__ void scatter_lds_kernel(const int* __restrict__ seg_ids,
                                   const int* __restrict__ flat_adj,
                                   const int* __restrict__ flat_j_idx,
                                   const float* __restrict__ quadratic,
                                   uint32_t* __restrict__ cursor,
                                   uint2* __restrict__ edge_vw, int E, int K) {
    extern __shared__ uint32_t h_sc[];
    const int t = threadIdx.x;
    for (int i = t; i < K; i += blockDim.x) h_sc[i] = 0u;
    __syncthreads();
    for (int e = blockIdx.x * blockDim.x + t; e < E; e += gridDim.x * blockDim.x)
        atomicAdd(&h_sc[seg_ids[e]], 1u);
    __syncthreads();
    for (int i = t; i < K; i += blockDim.x) {
        uint32_t c = h_sc[i];
        if (c) h_sc[i] = atomicAdd(&cursor[i], c);  // bin -> block base
    }
    __syncthreads();
    for (int e = blockIdx.x * blockDim.x + t; e < E; e += gridDim.x * blockDim.x) {
        int k = seg_ids[e];
        uint32_t pos = atomicAdd(&h_sc[k], 1u);
        edge_vw[pos] = make_uint2((uint32_t)flat_adj[e],
                                  __float_as_uint(quadratic[flat_j_idx[e]]));
    }
}

// One block per hidden unit; 128 threads = 128 batch elements = 4 wave32s.
// Edge tiles are double-buffered into LDS via async DMA. Within a wave the
// bit-word index (t>>5) and node index are uniform -> broadcast L2 hits.
__global__ __launch_bounds__(128) void rbm_field_kernel(
    const uint32_t* __restrict__ row_ptr,
    const uint2*    __restrict__ edge_vw,
    const uint32_t* __restrict__ s_bits,
    const float*    __restrict__ linear,
    const int*      __restrict__ hidden_idx,
    float*          __restrict__ out, int K, int NW) {
    const int k    = blockIdx.x;
    const int t    = threadIdx.x;
    const int lane = t & 31;   // batch bit within word
    const int word = t >> 5;   // which 32-bit word of the 128-bit mask

    const uint32_t start = row_ptr[k];
    const uint32_t end   = row_ptr[k + 1];

    __shared__ uint2 buf[2][EDGE_TILE];

    float acc = 0.0f;
    const uint32_t len = end - start;
    const uint32_t nt  = (len + EDGE_TILE - 1) / EDGE_TILE;

    if (nt) {
        // Prologue: async-stage tile 0 (each thread DMAs one uint2).
        {
            uint32_t idx = start + (uint32_t)t;
            if (idx < end) {
                uint32_t lds = (uint32_t)(uintptr_t)&buf[0][t];
                async_load_edge(lds, idx * 8u, edge_vw);
            }
        }
        int p = 0;
        for (uint32_t ti = 0; ti < nt; ++ti) {
            const uint32_t tb = start + ti * EDGE_TILE;
            if (ti + 1 < nt) {
                // Issue next tile into the alternate buffer, then wait until
                // only that newest DMA is outstanding (async ops complete in
                // order, so tile ti has landed).
                uint32_t idx = tb + EDGE_TILE + (uint32_t)t;
                if (idx < end) {
                    uint32_t lds = (uint32_t)(uintptr_t)&buf[p ^ 1][t];
                    async_load_edge(lds, idx * 8u, edge_vw);
                }
                async_wait_le1();
            } else {
                async_wait_0();
            }
            __syncthreads();   // tile ti visible to all 4 waves

            uint32_t n = end - tb;
            if (n > EDGE_TILE) n = EDGE_TILE;
            #pragma unroll 4
            for (uint32_t i = 0; i < n; ++i) {
                const uint2    e    = buf[p][i];
                const uint32_t bits = s_bits[(size_t)e.x * NW + word];
                // bit==1 -> +w, bit==0 -> -w : conditional sign-bit flip.
                const uint32_t neg  = ((~bits >> lane) & 1u) << 31;
                acc += __uint_as_float(e.y ^ neg);
            }
            __syncthreads();   // safe to overwrite buf[p] next iter
            p ^= 1;
        }
    }

    const float eff = acc + linear[hidden_idx[k]];
    out[(size_t)t * K + k] = tanhf(-eff);
}

extern "C" void kernel_launch(void* const* d_in, const int* in_sizes, int n_in,
                              void* d_out, int out_size, void* d_ws, size_t ws_size,
                              hipStream_t stream) {
    const float* linear     = (const float*)d_in[0];
    const float* quadratic  = (const float*)d_in[1];
    const int*   s          = (const int*)d_in[2];
    const int*   flat_adj   = (const int*)d_in[3];
    const int*   flat_j_idx = (const int*)d_in[4];
    const int*   seg_ids    = (const int*)d_in[5];
    const int*   hidden_idx = (const int*)d_in[6];

    const int N  = in_sizes[0];          // 55000 nodes
    const int E  = in_sizes[3];          // 1,000,000 edges
    const int K  = in_sizes[6];          // 5000 hidden units
    const int B  = in_sizes[2] / N;      // 128 batch
    const int NW = B / 32;               // 4 words per node

    // Workspace layout (~9 MB), each region 16B aligned.
    char*  ws  = (char*)d_ws;
    size_t off = 0;
    auto align16 = [](size_t x) { return (x + 15) & ~(size_t)15; };
    uint32_t* s_bits  = (uint32_t*)(ws + off); off = align16(off + (size_t)N * NW * 4);
    uint32_t* row_ptr = (uint32_t*)(ws + off); off = align16(off + (size_t)(K + 1) * 4);
    uint32_t* cursor  = (uint32_t*)(ws + off); off = align16(off + (size_t)K * 4);
    uint2*    edge_vw = (uint2*)(ws + off);    off = align16(off + (size_t)E * 8);
    (void)ws_size; (void)n_in; (void)out_size;

    const size_t hist_lds = (size_t)K * sizeof(uint32_t);

    // 1) zero histogram (cursor doubles as counts)
    zero_u32_kernel<<<(K + 255) / 256, 256, 0, stream>>>(cursor, K);
    // 2) bit-pack spins
    pack_bits_kernel<<<(N + 255) / 256, 256, 0, stream>>>(s, s_bits, N, NW);
    // 3) per-segment edge counts (LDS-aggregated)
    count_lds_kernel<<<CSR_BLOCKS, 1024, hist_lds, stream>>>(seg_ids, cursor, E, K);
    // 4) exclusive prefix scan -> CSR row pointers (single-wave WMMA scan)
    scan_counts_wmma_kernel<<<1, 32, 0, stream>>>(cursor, row_ptr, K);
    // 5) cursors = row starts
    init_cursor_kernel<<<(K + 255) / 256, 256, 0, stream>>>(row_ptr, cursor, K);
    // 6) scatter edges into CSR order (LDS range reservation)
    scatter_lds_kernel<<<CSR_BLOCKS, 1024, hist_lds, stream>>>(
        seg_ids, flat_adj, flat_j_idx, quadratic, cursor, edge_vw, E, K);
    // 7) main accumulation + tanh epilogue: one block per hidden unit
    rbm_field_kernel<<<K, B, 0, stream>>>(row_ptr, edge_vw, s_bits,
                                          linear, hidden_idx, (float*)d_out, K, NW);
}
